// EnsembleAdaptor_87136296501353
// MI455X (gfx1250) — matrix operand order repaired and checked by reading, not compiled, not run on hardware
//
#include <hip/hip_runtime.h>
#include <stdint.h>

typedef __attribute__((ext_vector_type(16))) __bf16 v16bf;
typedef __attribute__((ext_vector_type(2)))  __bf16 v2bf;
typedef __attribute__((ext_vector_type(8)))  float  v8f;
typedef __attribute__((ext_vector_type(4)))  unsigned int tdm_g0_t;
typedef __attribute__((ext_vector_type(8)))  int          tdm_g1_t;
typedef __attribute__((ext_vector_type(4)))  int          tdm_g2_t;

static constexpr int kB = 32, kS = 512, kDIN = 1024, kH = 4096, kDOUT = 1024;
static constexpr long long kNW1   = (long long)kH * kDIN;
static constexpr long long kNB1   = kH;
static constexpr long long kNW2   = (long long)kDOUT * kH;
static constexpr long long kTOTAL = kNW1 + kNB1 + kNW2 + kDOUT;

// hardware packed bf16 convert: one v_cvt_pk_bf16_f32 per pair
__device__ __forceinline__ uint32_t bf16_2(float lo, float hi) {
  v2bf v = { (__bf16)lo, (__bf16)hi };
  return __builtin_bit_cast(uint32_t, v);
}
__device__ __forceinline__ unsigned short bf16_s(float f) {
  __bf16 h = (__bf16)f;
  return __builtin_bit_cast(unsigned short, h);
}

struct B32 { uint4 lo, hi; };
__device__ __forceinline__ v16bf mkfrag(uint4 lo, uint4 hi) {
  B32 t{lo, hi};
  return __builtin_bit_cast(v16bf, t);
}

// ---- fp32 -> bf16 conversion of x into workspace ------------------------
__global__ void cvt_f32_bf16(const float* __restrict__ src,
                             unsigned short* __restrict__ dst,
                             long long n) {
  long long i = ((long long)blockIdx.x * blockDim.x + threadIdx.x) * 8;
  if (i >= n) return;
  float4 f0 = *(const float4*)(src + i);
  float4 f1 = *(const float4*)(src + i + 4);
  uint4 o;
  o.x = bf16_2(f0.x, f0.y);
  o.y = bf16_2(f0.z, f0.w);
  o.z = bf16_2(f1.x, f1.y);
  o.w = bf16_2(f1.z, f1.w);
  *(uint4*)(dst + i) = o;
}

// ---- tiled ensemble GEMM: out[b,s,n] = act(sum_k A[b,s,k]*W[b,n,k] + bias[b,n])
// Block tile 128(M) x 128(N) x 32(K); 8 wave32 waves, each 64x32.
// A-tile (bf16) staged by the Tensor Data Mover; W-tile converted fp32->bf16.
template <int K, bool RELU, bool OUT_BF16>
__global__ __launch_bounds__(256) void mlp_gemm(
    const unsigned short* __restrict__ Abase,  // bf16 [kB][kS][K]
    const float* __restrict__ Wall,            // ensemble weight blob (fp32)
    long long w_off, long long bias_off, int N,
    unsigned short* __restrict__ outb,         // bf16 output (layer 1)
    float* __restrict__ outf)                  // fp32 output (layer 2)
{
  constexpr int BK = 32;
  constexpr int PITCH = 40;                    // 32 elem + 16B pad (TDM pad too)
  constexpr int NK = K / BK;
  __shared__ unsigned short Asl[2][128 * PITCH];
  __shared__ unsigned short Bsl[2][128 * PITCH];

  const int b   = blockIdx.z;
  const int sb  = blockIdx.y * 128;
  const int nb  = blockIdx.x * 128;
  const int tid = threadIdx.x;
  const int lane = tid & 31;
  const int wave = tid >> 5;
  const int wm  = (wave >> 2) * 64;            // wave row offset (0 or 64)
  const int wn  = (wave & 3) * 32;             // wave col offset (0..96)
  const int half = lane >> 4;
  const int l16  = lane & 15;

  const unsigned short* A = Abase + (long long)b * kS * K + (long long)sb * K;
  const float* W    = Wall + (long long)b * kTOTAL + w_off + (long long)nb * K;
  const float* bias = Wall + (long long)b * kTOTAL + bias_off + nb;

  // ---- TDM: DMA the 128x32 bf16 A-tile into LDS with pitch-40 padding ----
  // D# group0: count=1 | lds_addr | global_addr | type=2
  // D# group1: data_size=2B, pad_enable, pad_interval=16dw, pad_amount=4dw,
  //            tensor_dim0=32, tensor_dim1=128, tile 32x128, stride0=K
  auto tdmA = [&](int kt, int st) {
    unsigned lds = (unsigned)(uintptr_t)(void*)&Asl[st][0];
    unsigned long long ga =
        (unsigned long long)(uintptr_t)(A + (long long)kt * BK);
    tdm_g0_t g0 = { 1u, lds, (unsigned)ga,
                    (unsigned)(ga >> 32) | 0x80000000u };
    tdm_g1_t g1 = { (int)((1u << 16) | (1u << 20) | (3u << 22) | (3u << 25)),
                    (int)(32u << 16),        // tensor_dim0[15:0] @ bits 63:48
                    (int)(128u << 16),       // dim0 hi=0 | tensor_dim1[15:0]
                    (int)(32u << 16),        // dim1 hi=0 | tile_dim0=32
                    (int)128,                // tile_dim1=128, tile_dim2=0
                    (int)K,                  // tensor_dim0_stride lo (elements)
                    0, 0 };
    tdm_g2_t gz  = { 0, 0, 0, 0 };
    tdm_g1_t gz8 = { 0, 0, 0, 0, 0, 0, 0, 0 };
    __builtin_amdgcn_tensor_load_to_lds(g0, g1, gz, gz, gz8, 0);
  };

  // cooperative W staging: thread t -> row t/2, 16-float chunk (t&1)*16
  const int ra = tid >> 1;
  const int ca = (tid & 1) * 16;

  v8f acc[4][2] = {};
  float4 w0, w1, w2, w3;

  auto gloadB = [&](int kt) {
    const float* wp = W + (long long)ra * K + kt * BK + ca;
    w0 = *(const float4*)(wp);
    w1 = *(const float4*)(wp + 4);
    w2 = *(const float4*)(wp + 8);
    w3 = *(const float4*)(wp + 12);
  };
  auto sstoreB = [&](int st) {
    uint4 p0, p1;
    p0.x = bf16_2(w0.x, w0.y);  p0.y = bf16_2(w0.z, w0.w);
    p0.z = bf16_2(w1.x, w1.y);  p0.w = bf16_2(w1.z, w1.w);
    p1.x = bf16_2(w2.x, w2.y);  p1.y = bf16_2(w2.z, w2.w);
    p1.z = bf16_2(w3.x, w3.y);  p1.w = bf16_2(w3.z, w3.w);
    unsigned short* bd = &Bsl[st][ra * PITCH + ca];
    *(uint4*)(bd)     = p0;
    *(uint4*)(bd + 8) = p1;
  };

  if (wave == 0) tdmA(0, 0);
  gloadB(0);
  sstoreB(0);
  if (wave == 0) __builtin_amdgcn_s_wait_tensorcnt(0);
  __syncthreads();

  for (int kt = 0; kt < NK; ++kt) {
    const int cur = kt & 1;
    const bool more = (kt + 1) < NK;
    if (more) {
      if (wave == 0) tdmA(kt + 1, 1 - cur);
      gloadB(kt + 1);
    }

    // B fragment (ISA: lane = col N=lane%16; K chunks at 8*half and 16+8*half)
    v16bf af[4], bfr[2];
#pragma unroll
    for (int ni = 0; ni < 2; ++ni) {
      const unsigned short* p = &Bsl[cur][(wn + ni * 16 + l16) * PITCH + 8 * half];
      bfr[ni] = mkfrag(*(const uint4*)(p), *(const uint4*)(p + 16));
    }
    // A fragment (ISA: lane = row M=lane%16; K chunks at 8*half and 16+8*half)
#pragma unroll
    for (int mi = 0; mi < 4; ++mi) {
      const unsigned short* p = &Asl[cur][(wm + mi * 16 + l16) * PITCH + 8 * half];
      af[mi] = mkfrag(*(const uint4*)(p), *(const uint4*)(p + 16));
    }
#pragma unroll
    for (int mi = 0; mi < 4; ++mi)
#pragma unroll
      for (int ni = 0; ni < 2; ++ni)
        acc[mi][ni] = __builtin_amdgcn_wmma_f32_16x16x32_bf16(
            false, af[mi], false, bfr[ni], (short)0, acc[mi][ni], false, false);

    if (more) sstoreB(1 - cur);
    if (more && wave == 0) __builtin_amdgcn_s_wait_tensorcnt(0);
    __syncthreads();
  }

  // epilogue: bias + relu + store (C/D layout: VGPR v -> row v + 8*half, col l16)
#pragma unroll
  for (int ni = 0; ni < 2; ++ni) {
    const int n = nb + wn + ni * 16 + l16;
    const float bv = bias[wn + ni * 16 + l16];
#pragma unroll
    for (int mi = 0; mi < 4; ++mi) {
#pragma unroll
      for (int v = 0; v < 8; ++v) {
        const int s = sb + wm + mi * 16 + half * 8 + v;
        float val = acc[mi][ni][v] + bv;
        if (RELU) val = fmaxf(val, 0.0f);
        const long long idx = (long long)b * kS * N + (long long)s * N + n;
        if (OUT_BF16) outb[idx] = bf16_s(val);
        else          outf[idx] = val;
      }
    }
  }
}

extern "C" void kernel_launch(void* const* d_in, const int* in_sizes, int n_in,
                              void* d_out, int out_size, void* d_ws, size_t ws_size,
                              hipStream_t stream) {
  const float* x = (const float*)d_in[0];
  const float* w = (const float*)d_in[1];
  float* y = (float*)d_out;

  unsigned short* X16 = (unsigned short*)d_ws;                 // 33.6 MB
  unsigned short* H16 = X16 + (size_t)kB * kS * kDIN;          // 134.2 MB

  const long long nx = (long long)kB * kS * kDIN;
  cvt_f32_bf16<<<dim3((unsigned)(nx / 8 / 256)), dim3(256), 0, stream>>>(x, X16, nx);

  // layer 1: h = relu(x @ W1^T + b1)  -> bf16 workspace
  mlp_gemm<kDIN, true, true>
      <<<dim3(kH / 128, kS / 128, kB), dim3(256), 0, stream>>>(
          X16, w, 0, kNW1, kH, H16, nullptr);

  // layer 2: y = h @ W2^T + b2 -> fp32 out
  mlp_gemm<kH, false, false>
      <<<dim3(kDOUT / 128, kS / 128, kB), dim3(256), 0, stream>>>(
          H16, w, kNW1 + kNB1, kNW1 + kNB1 + kNW2, kDOUT, nullptr, y);
}
